// CombinedModel_80659485818950
// MI455X (gfx1250) — matrix operand order repaired and checked by reading, not compiled, not run on hardware
//
#include <hip/hip_runtime.h>

typedef _Float16 h16;
typedef __attribute__((ext_vector_type(16))) _Float16 v16h;
typedef __attribute__((ext_vector_type(8)))  float    v8f;

union V16H { v16h v; uint4 q[2]; };

// A-tile (16x32 f16, M x K): lane holds row m = m0 + lane%16,
// K = kOff + (lane/16)*8 + {0..7} in VGPR0-3 and +16 + {0..7} in VGPR4-7.
__device__ inline v16h load_a_tile(const h16* base, int stride, int m0, int kOff,
                                   int lm, int lh) {
    const h16* p = base + (m0 + lm) * stride + kOff + lh * 8;
    V16H r;
    r.q[0] = *(const uint4*)(p);        // 8 halves: K kb..kb+7
    r.q[1] = *(const uint4*)(p + 16);   // 8 halves: K kb+16..kb+23
    return r.v;
}

// B-tile (32x16 f16, K x N) where B[k][n] = Wrow[n][k] (W stored row-major [n][k]):
// lane holds col n = n0 + lane%16, K = kOff + (lane/16)*8 + {0..7} and +16 + {0..7}.
__device__ inline v16h load_b_tile(const h16* W, int Kdim, int n0, int kOff,
                                   int lm, int lh) {
    const h16* p = W + (n0 + lm) * Kdim + kOff + lh * 8;
    V16H r;
    r.q[0] = *(const uint4*)(p);        // 8 halves
    r.q[1] = *(const uint4*)(p + 16);   // 8 halves: K kb+16..kb+23
    return r.v;
}

__device__ inline v8f wmma_f16(v16h a, v16h b, v8f c) {
    return __builtin_amdgcn_wmma_f32_16x16x32_f16(
        /*neg_a=*/false, a, /*neg_b=*/false, b,
        /*c_mod=*/(short)0, c, /*reuse_a=*/false, /*reuse_b=*/false);
}

// ---------------- prologue: convert weights to f16 into d_ws ----------------
// d_ws layout (halves): [0,2048) W1 64x32 ; [2048,6144) W2 64x64 ; [6144,7168) W3p 16x64
// floats at byte offset 16384: [0,64) b1 ; [64,128) b2 ; [128,144) b3 pad ;
//                              [144,160) mu pad ; [160,176) std pad
__global__ void prep_weights_k(const float* __restrict__ W1, const float* __restrict__ b1,
                               const float* __restrict__ W2, const float* __restrict__ b2,
                               const float* __restrict__ W3, const float* __restrict__ b3,
                               const float* __restrict__ mu, const float* __restrict__ sd,
                               h16* __restrict__ wh, float* __restrict__ fb) {
    const int t = threadIdx.x;
    for (int i = t; i < 64 * 32; i += 256) wh[i]        = (h16)W1[i];
    for (int i = t; i < 64 * 64; i += 256) wh[2048 + i] = (h16)W2[i];
    for (int i = t; i < 16 * 64; i += 256) {
        int r = i >> 6, c = i & 63;
        wh[6144 + i] = (h16)(r < 3 ? W3[r * 64 + c] : 0.0f);
    }
    if (t < 64)                    fb[t] = b1[t];
    else if (t < 128)              fb[t] = b2[t - 64];
    else if (t < 144)              fb[t] = (t - 128) < 3 ? b3[t - 128] : 0.0f;
    else if (t < 160)              fb[t] = (t - 144) < 3 ? mu[t - 144] : 0.0f;
    else if (t < 176)              fb[t] = (t - 160) < 3 ? sd[t - 160] : 1.0f;
}

// ---------------- fused gather + latent + WMMA MLP ----------------
__global__ __launch_bounds__(256) void neural_field_k(
    const int*   __restrict__ queries,       // [B,2]
    const int*   __restrict__ neighbor_map,  // [1024,1024,4]
    const float* __restrict__ positions,     // [N,2]
    const float* __restrict__ embeddings,    // [N,32]
    const h16*   __restrict__ wh,
    const float* __restrict__ fb,
    float*       __restrict__ out,           // [B,3]
    int nq)
{
    __shared__ __align__(16) h16 latLDS[256 * 32];  // 16 KB
    __shared__ __align__(16) h16 hLDS[256 * 64];    // 32 KB

    const int t = threadIdx.x;
    const int qbase = blockIdx.x * 256;

    // ---------------- Phase 1: gather + distances + latent (per thread) ----
    {
        int qi = qbase + t;
        if (qi >= nq) qi = nq - 1;   // tail clamp (stores guarded later)
        const int qr = queries[2 * qi];
        const int qc = queries[2 * qi + 1];
        const int4 nb = *(const int4*)(neighbor_map + (((qr << 10) + qc) << 2));
        const float qx = (float)qr, qy = (float)qc;

        float2 p0 = ((const float2*)positions)[nb.x];
        float2 p1 = ((const float2*)positions)[nb.y];
        float2 p2 = ((const float2*)positions)[nb.z];
        float2 p3 = ((const float2*)positions)[nb.w];
        const float d0 = sqrtf((p0.x - qx) * (p0.x - qx) + (p0.y - qy) * (p0.y - qy));
        const float d1 = sqrtf((p1.x - qx) * (p1.x - qx) + (p1.y - qy) * (p1.y - qy));
        const float d2 = sqrtf((p2.x - qx) * (p2.x - qx) + (p2.y - qy) * (p2.y - qy));
        const float d3 = sqrtf((p3.x - qx) * (p3.x - qx) + (p3.y - qy) * (p3.y - qy));

        const float4* e0 = (const float4*)(embeddings + (size_t)nb.x * 32);
        const float4* e1 = (const float4*)(embeddings + (size_t)nb.y * 32);
        const float4* e2 = (const float4*)(embeddings + (size_t)nb.z * 32);
        const float4* e3 = (const float4*)(embeddings + (size_t)nb.w * 32);

        #pragma unroll
        for (int c = 0; c < 8; ++c) {
            float4 a = e0[c], b = e1[c], cc = e2[c], dd = e3[c];
            union { h16 h[4]; uint2 u; } pk;
            pk.h[0] = (h16)(d0 * a.x + d1 * b.x + d2 * cc.x + d3 * dd.x);
            pk.h[1] = (h16)(d0 * a.y + d1 * b.y + d2 * cc.y + d3 * dd.y);
            pk.h[2] = (h16)(d0 * a.z + d1 * b.z + d2 * cc.z + d3 * dd.z);
            pk.h[3] = (h16)(d0 * a.w + d1 * b.w + d2 * cc.w + d3 * dd.w);
            *((uint2*)(latLDS + t * 32 + c * 4)) = pk.u;
        }
    }
    __syncthreads();

    // ---------------- Phase 2: WMMA MLP (per wave, wave-private rows) ------
    const int wave = t >> 5, lane = t & 31;
    const int lm = lane & 15, lh = lane >> 4;

    const h16* W1h = wh;          // 64x32
    const h16* W2h = wh + 2048;   // 64x64
    const h16* W3h = wh + 6144;   // 16x64 (rows 3..15 zero)
    const float* b1v = fb;
    const float* b2v = fb + 64;
    const float* b3v = fb + 128;
    const float* muv = fb + 144;
    const float* sdv = fb + 160;

    #pragma unroll
    for (int rt = 0; rt < 2; ++rt) {
        const int m0 = wave * 32 + rt * 16;

        // ---- Layer 1: latent(16x32) @ W1^T -> h1(16x64), relu ----
        const v16h a1 = load_a_tile(latLDS, 32, m0, 0, lm, lh);
        #pragma unroll
        for (int nt = 0; nt < 4; ++nt) {
            v8f c = {};
            c = wmma_f16(a1, load_b_tile(W1h, 32, nt * 16, 0, lm, lh), c);
            const float bb = b1v[nt * 16 + lm];
            #pragma unroll
            for (int r = 0; r < 8; ++r) {
                float v = c[r] + bb;
                v = v > 0.0f ? v : 0.0f;
                hLDS[(m0 + r + 8 * lh) * 64 + nt * 16 + lm] = (h16)v;
            }
        }

        // ---- Layer 2: h1(16x64) @ W2^T -> h2(16x64), relu (in-place rows) ----
        const v16h a20 = load_a_tile(hLDS, 64, m0, 0,  lm, lh);
        const v16h a21 = load_a_tile(hLDS, 64, m0, 32, lm, lh);
        #pragma unroll
        for (int nt = 0; nt < 4; ++nt) {
            v8f c = {};
            c = wmma_f16(a20, load_b_tile(W2h, 64, nt * 16, 0,  lm, lh), c);
            c = wmma_f16(a21, load_b_tile(W2h, 64, nt * 16, 32, lm, lh), c);
            const float bb = b2v[nt * 16 + lm];
            #pragma unroll
            for (int r = 0; r < 8; ++r) {
                float v = c[r] + bb;
                v = v > 0.0f ? v : 0.0f;
                hLDS[(m0 + r + 8 * lh) * 64 + nt * 16 + lm] = (h16)v;
            }
        }

        // ---- Layer 3: h2(16x64) @ W3p^T -> out(16x16), cols 0..2 valid ----
        const v16h a30 = load_a_tile(hLDS, 64, m0, 0,  lm, lh);
        const v16h a31 = load_a_tile(hLDS, 64, m0, 32, lm, lh);
        v8f c = {};
        c = wmma_f16(a30, load_b_tile(W3h, 64, 0, 0,  lm, lh), c);
        c = wmma_f16(a31, load_b_tile(W3h, 64, 0, 32, lm, lh), c);

        if (lm < 3) {
            const float bb = b3v[lm], sc = sdv[lm], mm = muv[lm];
            #pragma unroll
            for (int r = 0; r < 8; ++r) {
                const int m = m0 + r + 8 * lh;
                const int q = qbase + m;
                if (q < nq) {
                    float v = (c[r] + bb) * sc + mm;
                    v = fminf(fmaxf(v, 0.0f), 1.0f);
                    out[(size_t)q * 3 + lm] = v;
                }
            }
        }
    }
}

extern "C" void kernel_launch(void* const* d_in, const int* in_sizes, int n_in,
                              void* d_out, int out_size, void* d_ws, size_t ws_size,
                              hipStream_t stream) {
    const int*   queries      = (const int*)d_in[0];
    const int*   neighbor_map = (const int*)d_in[1];
    const float* positions    = (const float*)d_in[2];
    const float* embeddings   = (const float*)d_in[3];
    const float* W1 = (const float*)d_in[4];
    const float* b1 = (const float*)d_in[5];
    const float* W2 = (const float*)d_in[6];
    const float* b2 = (const float*)d_in[7];
    const float* W3 = (const float*)d_in[8];
    const float* b3 = (const float*)d_in[9];
    const float* mu = (const float*)d_in[10];
    const float* sd = (const float*)d_in[11];

    h16*   wh = (h16*)d_ws;
    float* fb = (float*)((char*)d_ws + 16384);

    prep_weights_k<<<1, 256, 0, stream>>>(W1, b1, W2, b2, W3, b3, mu, sd, wh, fb);

    const int nq = in_sizes[0] / 2;            // B
    const int blocks = (nq + 255) / 256;
    neural_field_k<<<blocks, 256, 0, stream>>>(queries, neighbor_map, positions,
                                               embeddings, wh, fb, (float*)d_out, nq);
}
